// XAIGuidedMultiheadAttention_17686675325377
// MI455X (gfx1250) — matrix operand order repaired and checked by reading, not compiled, not run on hardware
//
#include <hip/hip_runtime.h>
#include <hip/hip_bf16.h>

#define DIMD 512
#define NHEAD 8
#define HD 64
#define BB 2
#define SS 2048
#define MTOT (BB*SS)
#define WPB_ATT 8
#define KLS 72    // Klds row stride (elems): 64 + 8 pad (TDM pad 4 DW / 32 DW)
#define VLS 40    // Vlds row stride (elems): 32 + 8 pad (TDM pad 4 DW / 16 DW)
#define PSTR 40

typedef __attribute__((ext_vector_type(16))) __bf16 v16bf;
typedef __attribute__((ext_vector_type(8)))  float  v8f;
typedef __attribute__((ext_vector_type(8)))  __bf16 bf16x8;
typedef __attribute__((ext_vector_type(4)))  float  f32x4;
typedef __attribute__((ext_vector_type(4)))  unsigned int u32x4;
typedef __attribute__((ext_vector_type(8)))  int i32x8;
typedef __attribute__((ext_vector_type(4)))  int i32x4;

union AB16    { bf16x8 h[2]; v16bf v; };
union BF16x8u { __bf16 b[8];  bf16x8 v; };

__device__ __forceinline__ __bf16 f2bf(float f) { return static_cast<__bf16>(f); }

__device__ __forceinline__ void wavebar() {
#if defined(__HIP_DEVICE_COMPILE__)
  __builtin_amdgcn_wave_barrier();
#endif
}

// ---------------------------------------------------------------------------
// TDM: async 2-D tile load global -> LDS (D# per CDNA5 ISA ch.8).
// tile_d0 = contiguous elements per row, tile_d1 = rows, stride0 in elements,
// data_size = 2 bytes (bf16). LDS rows padded by (pad_amount+1) DWORDs every
// 2^(pad_interval+1) DWORDs.
// ---------------------------------------------------------------------------
__device__ __forceinline__ void tdm_load_2d(unsigned lds_addr, const void* gptr,
                                            unsigned tile_d0, unsigned tile_d1,
                                            unsigned long long stride0,
                                            unsigned pad_int, unsigned pad_amt) {
#if defined(__HIP_DEVICE_COMPILE__)
  unsigned long long ga = (unsigned long long)(uintptr_t)gptr;
  u32x4 g0;
  g0[0] = 1u;                                        // count=1, gather off
  g0[1] = lds_addr;                                  // LDS byte address
  g0[2] = (unsigned)(ga & 0xFFFFFFFFu);              // global_addr[31:0]
  g0[3] = (unsigned)((ga >> 32) & 0x1FFFFFFu) | (2u << 30);  // ga[56:32]|type=2
  const unsigned td0 = 0x100000u, td1 = 0x100000u;   // huge tensor dims (no OOB)
  i32x8 g1;
  g1[0] = (int)((1u << 16)                           // data_size = 2B
              | (1u << 20)                           // pad_enable
              | (pad_int << 22) | (pad_amt << 25));
  g1[1] = (int)((td0 & 0xFFFFu) << 16);              // barrier_addr=0 | dim0.lo
  g1[2] = (int)((td0 >> 16) | ((td1 & 0xFFFFu) << 16));      // dim0.hi|dim1.lo
  g1[3] = (int)((td1 >> 16) | (tile_d0 << 16));              // dim1.hi|tile0
  g1[4] = (int)tile_d1;                              // tile1 | tile2=0
  g1[5] = (int)(unsigned)(stride0 & 0xFFFFFFFFull);  // dim0_stride.lo
  g1[6] = (int)(unsigned)(stride0 >> 32);            // dim0_stride.hi|d1s.lo=0
  g1[7] = 0;
  i32x4 gz = 0;
#if __clang_major__ >= 23
  i32x8 gz8 = 0;
  __builtin_amdgcn_tensor_load_to_lds(g0, g1, gz, gz, gz8, 0);
#else
  __builtin_amdgcn_tensor_load_to_lds(g0, g1, gz, gz, 0);
#endif
#endif
}

template <int N>
__device__ __forceinline__ void wait_tensorcnt() {
#if defined(__HIP_DEVICE_COMPILE__)
  __builtin_amdgcn_s_wait_tensorcnt(N);    // immediate required
#endif
}

__device__ __forceinline__ unsigned lds_off(const void* p) {
  return (unsigned)(uintptr_t)p;   // low 32 bits of generic ptr = LDS offset
}

// ---------------------------------------------------------------------------
// Elementwise fp32 -> bf16 (one pass; all GEMMs read bf16). n % 8 == 0.
// ---------------------------------------------------------------------------
__global__ void cvt_bf16_kernel(const float* __restrict__ in,
                                __bf16* __restrict__ out, int n) {
  int i = (blockIdx.x * blockDim.x + threadIdx.x) * 8;
  if (i >= n) return;
  f32x4 a = *(const f32x4*)(in + i);
  f32x4 b = *(const f32x4*)(in + i + 4);
  BF16x8u r;
  #pragma unroll
  for (int j = 0; j < 4; ++j) { r.b[j] = f2bf(a[j]); r.b[4 + j] = f2bf(b[j]); }
  *(bf16x8*)(out + i) = r.v;
}

// ---------------------------------------------------------------------------
// C[M,N] = A[M,K] @ W[N,K]^T + bias ; bf16 in, fp32 acc, bf16 out
// (row-major, or transposed per-batch as Vt[b][n][s]). One wave / 16x16 tile.
// ---------------------------------------------------------------------------
__global__ void proj_gemm_kernel(const __bf16* __restrict__ A,
                                 const __bf16* __restrict__ W,
                                 const float* __restrict__ bias,
                                 __bf16* __restrict__ outR,
                                 __bf16* __restrict__ outT) {
  const int lane = threadIdx.x & 31;
  const int half = lane >> 4, lr = lane & 15;
  const int gw = blockIdx.x * (blockDim.x >> 5) + (threadIdx.x >> 5);
  const int TN = DIMD / 16;
  const int tn = gw % TN, tm = gw / TN;

  const __bf16* arow = A + (size_t)(tm * 16 + lr) * DIMD;
  const __bf16* wrow = W + (size_t)(tn * 16 + lr) * DIMD;

  v8f acc = 0.0f;
  #pragma unroll 4
  for (int kk = 0; kk < DIMD; kk += 32) {
    AB16 av, bv;
    av.h[0] = *(const bf16x8*)(arow + kk + half * 8);
    av.h[1] = *(const bf16x8*)(arow + kk + 16 + half * 8);
    bv.h[0] = *(const bf16x8*)(wrow + kk + half * 8);
    bv.h[1] = *(const bf16x8*)(wrow + kk + 16 + half * 8);
    acc = __builtin_amdgcn_wmma_f32_16x16x32_bf16(false, av.v, false, bv.v,
                                                  (short)0, acc, false, false);
  }
  const int nco = tn * 16 + lr;
  const float bn = bias[nco];
  if (outT) {                       // Vt[b][n][s] (tile never crosses batch)
    const int b = (tm * 16) / SS;
    const int s0 = tm * 16 + 8 * half - b * SS;
    BF16x8u pk;
    #pragma unroll
    for (int v = 0; v < 8; ++v) pk.b[v] = f2bf(acc[v] + bn);
    *(bf16x8*)(outT + ((size_t)b * DIMD + nco) * SS + s0) = pk.v;
  } else {
    #pragma unroll
    for (int v = 0; v < 8; ++v)
      outR[(size_t)(tm * 16 + v + 8 * half) * DIMD + nco] = f2bf(acc[v] + bn);
  }
}

// ---------------------------------------------------------------------------
// Flash attention. Block = 8 waves = 8 query tiles of one (b,h).
// K/V tiles are DMA'd into LDS by the Tensor Data Mover (double buffered,
// hardware row padding), all waves consume from LDS.
// ---------------------------------------------------------------------------
__global__ void attn_kernel(const __bf16* __restrict__ Qb,
                            const __bf16* __restrict__ Kb,
                            const __bf16* __restrict__ Vt,
                            __bf16* __restrict__ On) {
  __shared__ __align__(16) __bf16 Klds[2][32][KLS];   //  9216 B
  __shared__ __align__(16) __bf16 Vlds[2][HD][VLS];   // 10240 B
  __shared__ __align__(16) __bf16 Plds[WPB_ATT][16][PSTR];
  const int lane = threadIdx.x & 31;
  const int half = lane >> 4, lr = lane & 15;
  const int wl = threadIdx.x >> 5;
  const int gw = blockIdx.x * WPB_ATT + wl;
  const int QT = SS / 16;
  const int qt = gw % QT;
  const int hh = (gw / QT) % NHEAD;
  const int b  = gw / (QT * NHEAD);
  const int q0 = qt * 16;

  AB16 qa[2];                                     // Q 16x64 as two A-operands
  {
    const __bf16* qrow = Qb + ((size_t)(b * SS + q0 + lr)) * DIMD + hh * HD;
    #pragma unroll
    for (int c = 0; c < 2; ++c) {
      qa[c].h[0] = *(const bf16x8*)(qrow + c * 32 + half * 8);
      qa[c].h[1] = *(const bf16x8*)(qrow + c * 32 + 16 + half * 8);
    }
  }

  const __bf16* Kg = Kb + ((size_t)b * SS) * DIMD + hh * HD;   // + kb*DIMD
  const __bf16* Vg = Vt + ((size_t)b * DIMD + hh * HD) * SS;   // + kb

  float mst[8], lst[8];
  v8f acc[4];
  #pragma unroll
  for (int v = 0; v < 8; ++v) { mst[v] = -3.0e38f; lst[v] = 0.0f; }
  #pragma unroll
  for (int d = 0; d < 4; ++d) acc[d] = 0.0f;

  const float sc = 0.125f * 1.44269504089f;       // 1/sqrt(hd) * log2(e)
  const int nIter = SS / 32;                      // 64

  if (wl == 0) {                                  // prologue: tile 0
    tdm_load_2d(lds_off(&Klds[0][0][0]), Kg, HD, 32, DIMD, 4, 3);
    tdm_load_2d(lds_off(&Vlds[0][0][0]), Vg, 32, HD, SS,   3, 3);
  }

  for (int i = 0; i < nIter; ++i) {
    const int kb = i * 32, buf = i & 1;
    if (wl == 0) {
      if (i + 1 < nIter) {        // issue next pair, then wait for current
        tdm_load_2d(lds_off(&Klds[buf ^ 1][0][0]), Kg + (size_t)(kb + 32) * DIMD,
                    HD, 32, DIMD, 4, 3);
        tdm_load_2d(lds_off(&Vlds[buf ^ 1][0][0]), Vg + kb + 32,
                    32, HD, SS, 3, 3);
        wait_tensorcnt<2>();      // in-order: current pair landed
      } else {
        wait_tensorcnt<0>();
      }
    }
    __syncthreads();              // publish buf to all waves

    // ---- scores S = Q @ K^T for two 16-key tiles (from LDS) ----
    v8f s0 = 0.0f, s1 = 0.0f;
    #pragma unroll
    for (int c = 0; c < 2; ++c) {
      const __bf16* kp0 = &Klds[buf][lr][c * 32 + half * 8];
      AB16 b0; b0.h[0] = *(const bf16x8*)(kp0); b0.h[1] = *(const bf16x8*)(kp0 + 16);
      s0 = __builtin_amdgcn_wmma_f32_16x16x32_bf16(false, qa[c].v, false, b0.v,
                                                   (short)0, s0, false, false);
      const __bf16* kp1 = &Klds[buf][16 + lr][c * 32 + half * 8];
      AB16 b1; b1.h[0] = *(const bf16x8*)(kp1); b1.h[1] = *(const bf16x8*)(kp1 + 16);
      s1 = __builtin_amdgcn_wmma_f32_16x16x32_bf16(false, qa[c].v, false, b1.v,
                                                   (short)0, s1, false, false);
    }
    // ---- online softmax (base-2); row m = v + 8*half ----
    float p0[8], p1[8];
    #pragma unroll
    for (int v = 0; v < 8; ++v) {
      float t0 = s0[v] * sc, t1 = s1[v] * sc;
      float cm = fmaxf(t0, t1);
      cm = fmaxf(cm, __shfl_xor(cm, 1, 32));
      cm = fmaxf(cm, __shfl_xor(cm, 2, 32));
      cm = fmaxf(cm, __shfl_xor(cm, 4, 32));
      cm = fmaxf(cm, __shfl_xor(cm, 8, 32));
      float mn = fmaxf(mst[v], cm);
      float al = exp2f(mst[v] - mn);
      mst[v] = mn;
      p0[v] = exp2f(t0 - mn);
      p1[v] = exp2f(t1 - mn);
      float rs = p0[v] + p1[v];
      rs += __shfl_xor(rs, 1, 32);
      rs += __shfl_xor(rs, 2, 32);
      rs += __shfl_xor(rs, 4, 32);
      rs += __shfl_xor(rs, 8, 32);
      lst[v] = lst[v] * al + rs;
      acc[0][v] *= al; acc[1][v] *= al; acc[2][v] *= al; acc[3][v] *= al;
    }
    // ---- P: C-layout -> A-layout through wave-private LDS ----
    #pragma unroll
    for (int v = 0; v < 8; ++v) {
      Plds[wl][v + 8 * half][lr]      = f2bf(p0[v]);
      Plds[wl][v + 8 * half][16 + lr] = f2bf(p1[v]);
    }
    wavebar();
    AB16 pa;
    pa.h[0] = *(const bf16x8*)(&Plds[wl][lr][half * 8]);
    pa.h[1] = *(const bf16x8*)(&Plds[wl][lr][16 + half * 8]);
    wavebar();
    // ---- O += P @ V (V tile transposed in LDS) ----
    #pragma unroll
    for (int dt = 0; dt < 4; ++dt) {
      const __bf16* vp = &Vlds[buf][dt * 16 + lr][half * 8];
      AB16 vb; vb.h[0] = *(const bf16x8*)(vp); vb.h[1] = *(const bf16x8*)(vp + 16);
      acc[dt] = __builtin_amdgcn_wmma_f32_16x16x32_bf16(false, pa.v, false, vb.v,
                                                        (short)0, acc[dt], false, false);
    }
    __syncthreads();              // everyone done with buf before it's reloaded
  }

  // ---- normalize and store bf16 row-major [B,S,D] ----
  #pragma unroll
  for (int v = 0; v < 8; ++v) {
    float inv = 1.0f / lst[v];
    __bf16* orow = On + ((size_t)(b * SS + q0 + v + 8 * half)) * DIMD + hh * HD;
    #pragma unroll
    for (int dt = 0; dt < 4; ++dt)
      orow[dt * 16 + lr] = f2bf(acc[dt][v] * inv);
  }
}

// ---------------------------------------------------------------------------
// out[M,N] = attn(bf16)[M,K] @ Wo(bf16)[N,K]^T + bo ; fp32 output.
// ---------------------------------------------------------------------------
__global__ void outproj_kernel(const __bf16* __restrict__ A,
                               const __bf16* __restrict__ W,
                               const float* __restrict__ bias,
                               float* __restrict__ out) {
  const int lane = threadIdx.x & 31;
  const int half = lane >> 4, lr = lane & 15;
  const int gw = blockIdx.x * (blockDim.x >> 5) + (threadIdx.x >> 5);
  const int TN = DIMD / 16;
  const int tn = gw % TN, tm = gw / TN;

  const __bf16* arow = A + (size_t)(tm * 16 + lr) * DIMD;
  const __bf16* wrow = W + (size_t)(tn * 16 + lr) * DIMD;

  v8f acc = 0.0f;
  #pragma unroll 4
  for (int kk = 0; kk < DIMD; kk += 32) {
    AB16 av, bv;
    av.h[0] = *(const bf16x8*)(arow + kk + half * 8);
    av.h[1] = *(const bf16x8*)(arow + kk + 16 + half * 8);
    bv.h[0] = *(const bf16x8*)(wrow + kk + half * 8);
    bv.h[1] = *(const bf16x8*)(wrow + kk + 16 + half * 8);
    acc = __builtin_amdgcn_wmma_f32_16x16x32_bf16(false, av.v, false, bv.v,
                                                  (short)0, acc, false, false);
  }
  const int n = tn * 16 + lr;
  const float bn = bias[n];
  #pragma unroll
  for (int v = 0; v < 8; ++v)
    out[(size_t)(tm * 16 + v + 8 * half) * DIMD + n] = acc[v] + bn;
}

// ---------------------------------------------------------------------------
extern "C" void kernel_launch(void* const* d_in, const int* in_sizes, int n_in,
                              void* d_out, int out_size, void* d_ws, size_t ws_size,
                              hipStream_t stream) {
  (void)in_sizes; (void)n_in; (void)out_size; (void)ws_size;
  const float* x     = (const float*)d_in[0];
  const float* Qexpl = (const float*)d_in[1];
  const float* Kexpl = (const float*)d_in[2];
  const float* Wq = (const float*)d_in[7];
  const float* bq = (const float*)d_in[8];
  const float* Wk = (const float*)d_in[9];
  const float* bk = (const float*)d_in[10];
  const float* Wv = (const float*)d_in[11];
  const float* bv = (const float*)d_in[12];
  const float* Wo = (const float*)d_in[13];
  const float* bo = (const float*)d_in[14];

  const size_t MD = (size_t)MTOT * DIMD;        // 2,097,152 elements
  const size_t WD = (size_t)DIMD * DIMD;        //   262,144 elements
  __bf16* Qb  = (__bf16*)d_ws;    // [B*S, D]  Q projection (bf16)
  __bf16* Kb  = Qb  + MD;         // [B*S, D]  K projection
  __bf16* Vt  = Kb  + MD;         // [B, D, S] V projection, transposed
  __bf16* On  = Vt  + MD;         // [B*S, D]  attention output
  __bf16* Xb  = On  + MD;         // bf16 copies of fp32 inputs
  __bf16* Qeb = Xb  + MD;
  __bf16* Keb = Qeb + MD;
  __bf16* Wqb = Keb + MD;
  __bf16* Wkb = Wqb + WD;
  __bf16* Wvb = Wkb + WD;
  __bf16* Wob = Wvb + WD;

  dim3 blk(256);
  dim3 gCvtA((unsigned)(MD / 8 / 256));
  dim3 gCvtW((unsigned)(WD / 8 / 256));
  cvt_bf16_kernel<<<gCvtA, blk, 0, stream>>>(x,     Xb,  (int)MD);
  cvt_bf16_kernel<<<gCvtA, blk, 0, stream>>>(Qexpl, Qeb, (int)MD);
  cvt_bf16_kernel<<<gCvtA, blk, 0, stream>>>(Kexpl, Keb, (int)MD);
  cvt_bf16_kernel<<<gCvtW, blk, 0, stream>>>(Wq, Wqb, (int)WD);
  cvt_bf16_kernel<<<gCvtW, blk, 0, stream>>>(Wk, Wkb, (int)WD);
  cvt_bf16_kernel<<<gCvtW, blk, 0, stream>>>(Wv, Wvb, (int)WD);
  cvt_bf16_kernel<<<gCvtW, blk, 0, stream>>>(Wo, Wob, (int)WD);

  dim3 gGemm(((MTOT / 16) * (DIMD / 16)) / 8);          // 1024 blocks
  proj_gemm_kernel<<<gGemm, blk, 0, stream>>>(Qeb, Wqb, bq, Qb, nullptr);
  proj_gemm_kernel<<<gGemm, blk, 0, stream>>>(Keb, Wkb, bk, Kb, nullptr);
  proj_gemm_kernel<<<gGemm, blk, 0, stream>>>(Xb,  Wvb, bv, nullptr, Vt);
  dim3 gAtt((BB * NHEAD * (SS / 16)) / WPB_ATT);        // 256 blocks
  attn_kernel<<<gAtt, blk, 0, stream>>>(Qb, Kb, Vt, On);
  outproj_kernel<<<gGemm, blk, 0, stream>>>(On, Wob, bo, (float*)d_out);
}
